// sFastGuidedFilter_9096740733328
// MI455X (gfx1250) — matrix operand order repaired and checked by reading, not compiled, not run on hardware
//
#include <hip/hip_runtime.h>

typedef _Float16 v16h __attribute__((ext_vector_type(16)));
typedef float    v8f  __attribute__((ext_vector_type(8)));

#define LR   256
#define HR   1024
#define RAD  4
#define EPSF 2.0f

// ---------------------------------------------------------------------------
// Kernel 1: low-res guided filter coefficients A,b (9x9 replicated box filter)
// ---------------------------------------------------------------------------
__global__ __launch_bounds__(256) void gf_lowres(const float* __restrict__ lrx,
                                                 const float* __restrict__ lry,
                                                 float* __restrict__ Ac,
                                                 float* __restrict__ bc) {
  int idx = blockIdx.x * blockDim.x + threadIdx.x;
  if (idx >= 3 * LR * LR) return;
  int c = idx >> 16;               // LR*LR == 65536
  int rem = idx & 65535;
  int y = rem >> 8;
  int x = rem & 255;
  const float* px = lrx + c * LR * LR;
  const float* py = lry + c * LR * LR;
  float sx = 0.f, sy = 0.f, sxy = 0.f, sxx = 0.f;
  for (int dy = -RAD; dy <= RAD; ++dy) {
    int yy = min(max(y + dy, 0), LR - 1);
    const float* rx = px + yy * LR;
    const float* ry = py + yy * LR;
    for (int dx = -RAD; dx <= RAD; ++dx) {
      int xx = min(max(x + dx, 0), LR - 1);
      float gx = rx[xx], gy = ry[xx];
      sx += gx; sy += gy; sxy += gx * gy; sxx += gx * gx;
    }
  }
  const float inv81 = 1.0f / 81.0f;
  float mx = sx * inv81, my = sy * inv81, mxy = sxy * inv81, mxx = sxx * inv81;
  float Av = (mxy - mx * my) / ((mxx - mx * mx) + EPSF);
  float bv = my - Av * mx;
  Ac[idx] = Av;
  bc[idx] = bv;
}

// ---------------------------------------------------------------------------
// Kernel 2: upsample + 8-direction side-window stencil via WMMA banded matmul.
// One wave32 per 16x16 output tile per channel.
// ---------------------------------------------------------------------------
struct SmT {
  float imt[16][16];                      // HR guide tile (async-staged)
  float pA[8][8];                         // low-res A patch
  float pB[8][8];                         // low-res b patch
  float rIA[24][8];                       // row-interp at patch cols
  float rIB[24][8];
  float upA[24][24];                      // upsampled halo
  float upB[24][24];
  __align__(32) _Float16 S5A[20][32];     // transposed 5-tap col sums [jj][k]
  __align__(32) _Float16 S5B[20][32];
  __align__(32) _Float16 MTA[16][32];     // transposed center column [j][k]
  __align__(32) _Float16 MTB[16][32];
};

__global__ __launch_bounds__(32) void gf_highres(const float* __restrict__ hrx,
                                                 const float* __restrict__ Ac,
                                                 const float* __restrict__ bc,
                                                 float* __restrict__ out) {
  __shared__ SmT sm;

  int tile = blockIdx.x;
  int c  = tile / (64 * 64);
  int t2 = tile % (64 * 64);
  int ty = t2 / 64, tx = t2 % 64;
  int m0 = ty * 16, n0 = tx * 16;
  int lane = threadIdx.x;                 // 0..31 (wave32)

  const float* Ap  = Ac  + c * LR * LR;
  const float* Bp  = bc  + c * LR * LR;
  const float* imp = hrx + (size_t)c * HR * HR;

  // ---- async stage of the 16x16 HR guide tile into LDS (64x b128, 2/lane) ----
#pragma unroll
  for (int i = 0; i < 2; ++i) {
    int p = lane + i * 32;                // 0..63
    int r = p >> 2;
    int q = (p & 3) * 4;
    unsigned long long ga =
        (unsigned long long)(uintptr_t)(imp + (size_t)(m0 + r) * HR + n0 + q);
    unsigned ldsoff = (unsigned)((char*)&sm.imt[r][q] - (char*)&sm);
    asm volatile("global_load_async_to_lds_b128 %0, %1, off"
                 :: "v"(ldsoff), "v"(ga) : "memory");
  }

  // ---- stage the (<=8x8) low-res A,b footprint into LDS ----
  const float scale = 255.0f / 1023.0f;
  int iy_lo = (int)((float)max(m0 - 4, 0) * scale);
  int ix_lo = (int)((float)max(n0 - 4, 0) * scale);
#pragma unroll
  for (int i = 0; i < 4; ++i) {
    int p = lane + i * 32;                // 0..127
    int plane = p >> 6;
    int rr = (p >> 3) & 7, cc = p & 7;
    int iy = min(iy_lo + rr, LR - 1);
    int ix = min(ix_lo + cc, LR - 1);
    float v = (plane ? Bp : Ap)[iy * LR + ix];
    if (plane) sm.pB[rr][cc] = v; else sm.pA[rr][cc] = v;
  }
  __syncthreads();

  // ---- separable bilinear, pass 1: row lerp at patch columns (24x8) ----
  for (int p = lane; p < 384; p += 32) {
    int plane = p >= 192;
    int q = p & 191;
    int k = q >> 3, cc = q & 7;
    int yy = min(max(m0 - 4 + k, 0), HR - 1);
    float tyf = yy * scale; int iy = min((int)tyf, LR - 2); float fy = tyf - iy;
    int ry = iy - iy_lo;                      // 0..6 by construction
    if (plane) {
      float a = sm.pB[ry][cc];
      sm.rIB[k][cc] = a + fy * (sm.pB[ry + 1][cc] - a);
    } else {
      float a = sm.pA[ry][cc];
      sm.rIA[k][cc] = a + fy * (sm.pA[ry + 1][cc] - a);
    }
  }
  __syncthreads();

  // ---- separable bilinear, pass 2: column lerp to 24x24 halo ----
  for (int p = lane; p < 576; p += 32) {
    int k = p / 24, j = p % 24;
    int xx = min(max(n0 - 4 + j, 0), HR - 1);
    float txf = xx * scale; int ix = min((int)txf, LR - 2); float fx = txf - ix;
    int rx = ix - ix_lo;                      // 0..6 by construction
    float a0 = sm.rIA[k][rx];
    float b0 = sm.rIB[k][rx];
    sm.upA[k][j] = a0 + fx * (sm.rIA[k][rx + 1] - a0);
    sm.upB[k][j] = b0 + fx * (sm.rIB[k][rx + 1] - b0);
  }
  __syncthreads();

  // ---- shared 5-tap horizontal sums S5[k][jj], jj in [0,20) (SL=S5[j], SR=S5[j+4]) ----
  for (int p = lane; p < 480; p += 32) {    // 24*20
    int k = p / 20, jj = p % 20;
    float sa = sm.upA[k][jj], sb = sm.upB[k][jj];
#pragma unroll
    for (int t = 1; t < 5; ++t) { sa += sm.upA[k][jj + t]; sb += sm.upB[k][jj + t]; }
    sm.S5A[jj][k] = (_Float16)sa;
    sm.S5B[jj][k] = (_Float16)sb;
  }
  // center column planes (for SH = SL + SR - mid)
  for (int p = lane; p < 384; p += 32) {    // 24*16
    int k = p >> 4, j = p & 15;
    sm.MTA[j][k] = (_Float16)sm.upA[k][j + 4];
    sm.MTB[j][k] = (_Float16)sm.upB[k][j + 4];
  }
  // zero-pad K = 24..31
  for (int p = lane; p < 320; p += 32) {    // 2 * 20*8
    int plane = p >= 160; int q = p % 160;
    int jj = q >> 3, kk = 24 + (q & 7);
    if (plane) sm.S5B[jj][kk] = (_Float16)0.0f; else sm.S5A[jj][kk] = (_Float16)0.0f;
  }
  for (int p = lane; p < 256; p += 32) {    // 2 * 16*8
    int plane = p >= 128; int q = p % 128;
    int j = q >> 3, kk = 24 + (q & 7);
    if (plane) sm.MTB[j][kk] = (_Float16)0.0f; else sm.MTA[j][kk] = (_Float16)0.0f;
  }
  __builtin_amdgcn_s_wait_asynccnt(0);
  __syncthreads();

  // ---- per-lane operand fetch (WMMA layouts, wave32) ----
  int g  = lane >> 4;                     // half-wave group
  int nn = lane & 15;                     // column (B/C) and row M (A)
  int kb = g * 16;

  float im[8];
#pragma unroll
  for (int v = 0; v < 8; ++v) im[v] = sm.imt[v + 8 * g][nn];   // C layout rows

  // B operands: SL/SR from the same S5 plane; SH synthesized with packed f16 math
  v16h bop[6];
  bop[0] = *(const v16h*)&sm.S5A[nn][kb];          // SL_A
  bop[1] = *(const v16h*)&sm.S5A[nn + 4][kb];      // SR_A
  bop[3] = *(const v16h*)&sm.S5B[nn][kb];          // SL_b
  bop[4] = *(const v16h*)&sm.S5B[nn + 4][kb];      // SR_b
  {
    v16h mda = *(const v16h*)&sm.MTA[nn][kb];
    v16h mdb = *(const v16h*)&sm.MTB[nn][kb];
    bop[2] = bop[0] + bop[1] - mda;                // SH_A (v_pk f16)
    bop[5] = bop[3] + bop[4] - mdb;                // SH_b
  }

  // ---- build the 5 distinct banded A-weight matrices once ----
  // 0: full/45  1: top/45  2: bot/45  3: top/25  4: bot/25
  const int   cfg_lo[5] = {0, 0, 4, 0, 4};
  const int   cfg_hi[5] = {8, 4, 8, 4, 8};
  const float cfg_nm[5] = {1.f/45, 1.f/45, 1.f/45, 1.f/25, 1.f/25};
  int m = lane & 15;                      // A-matrix row held by this lane
  v16h awm[5];
#pragma unroll
  for (int w = 0; w < 5; ++w) {
    int klo = m + cfg_lo[w], khi = m + cfg_hi[w];
    float nrm = cfg_nm[w];
#pragma unroll
    for (int s = 0; s < 16; ++s) {
      int v = s >> 1, par = s & 1;
      int k = 16 * g + 2 * v + par;   // contiguous K layout (k = 16g + s), matches B
      awm[w][s] = (_Float16)((k >= klo && k <= khi) ? nrm : 0.f);
    }
  }

  // direction tables: L,R,U,D,NW,NE,SW,SE
  const int plidx[8] = {0, 1, 2, 2, 0, 1, 0, 1};
  const int awidx[8] = {0, 0, 1, 2, 3, 3, 4, 4};

  float bestd[8], bestad[8];
#pragma unroll
  for (int v = 0; v < 8; ++v) { bestd[v] = 0.f; bestad[v] = 3.4e38f; }

#pragma unroll
  for (int dir = 0; dir < 8; ++dir) {
    v8f zero = {};
    v8f cA = __builtin_amdgcn_wmma_f32_16x16x32_f16(
        false, awm[awidx[dir]], false, bop[plidx[dir]],     (short)0, zero, false, false);
    v8f cB = __builtin_amdgcn_wmma_f32_16x16x32_f16(
        false, awm[awidx[dir]], false, bop[plidx[dir] + 3], (short)0, zero, false, false);
#pragma unroll
    for (int v = 0; v < 8; ++v) {
      float d  = cA[v] * im[v] + cB[v] - im[v];
      float ad = fabsf(d);
      bool take = ad < bestad[v];
      bestad[v] = take ? ad : bestad[v];
      bestd[v]  = take ? d  : bestd[v];
    }
  }

  // ---- trunc + clamp + store ----
#pragma unroll
  for (int v = 0; v < 8; ++v) {
    float r = truncf(bestd[v] + im[v]);
    r = fminf(fmaxf(r, 0.f), 255.f);
    out[(size_t)c * HR * HR + (size_t)(m0 + v + 8 * g) * HR + (n0 + nn)] = r;
  }
}

// ---------------------------------------------------------------------------
extern "C" void kernel_launch(void* const* d_in, const int* in_sizes, int n_in,
                              void* d_out, int out_size, void* d_ws, size_t ws_size,
                              hipStream_t stream) {
  (void)in_sizes; (void)n_in; (void)out_size; (void)ws_size;
  const float* lrx = (const float*)d_in[0];
  const float* lry = (const float*)d_in[1];
  const float* hrx = (const float*)d_in[2];
  // d_in[3] = box_w (constant 1/81 depthwise weights; folded into kernel 1)

  float* Ac = (float*)d_ws;                    // 3*256*256 floats
  float* bc = Ac + 3 * LR * LR;                // 3*256*256 floats
  float* out = (float*)d_out;

  gf_lowres<<<(3 * LR * LR) / 256, 256, 0, stream>>>(lrx, lry, Ac, bc);
  gf_highres<<<64 * 64 * 3, 32, 0, stream>>>(hrx, Ac, bc, out);
}
